// TransLayer_52072183497304
// MI455X (gfx1250) — matrix operand order repaired and checked by reading, not compiled, not run on hardware
//
#include <hip/hip_runtime.h>

typedef __bf16 bf16_t;
typedef __bf16 v16bf __attribute__((ext_vector_type(16)));
typedef float  v8f   __attribute__((ext_vector_type(8)));
typedef float  v2f   __attribute__((ext_vector_type(2)));

// ---------------------------------------------------------------------------
// WMMA wrappers (CDNA5 / gfx1250, wave32)
// ---------------------------------------------------------------------------
__device__ inline v8f wmma_bf16(v16bf a, v16bf b, v8f c) {
  // (neg_a, A, neg_b, B, c_mod, C, reuse_a, reuse_b)
  return __builtin_amdgcn_wmma_f32_16x16x32_bf16(false, a, false, b, (short)0, c, false, false);
}
__device__ inline v8f wmma_f32(v2f a, v2f b, v8f c) {
  return __builtin_amdgcn_wmma_f32_16x16x4_f32(false, a, false, b, (short)0, c, false, false);
}

// ---------------------------------------------------------------------------
// Fragment loaders; bf16 A/B both use contiguous k = 16*half + e (not the ISA table order)
// ---------------------------------------------------------------------------
// A (16x32 bf16): lane<16 -> M=lane, K in {0..15}; lane>=16 -> K in {16..31}
__device__ inline v16bf load_a_frag(const bf16_t* base, int stride, int lane) {
  const int m = lane & 15;
  const int half = (lane >> 4) & 1;
  v16bf f;
#pragma unroll
  for (int e = 0; e < 16; ++e) {
    const int p = e >> 1, lo = e & 1;
    const int k = half * 16 + 2 * p + lo;
    f[e] = base[m * stride + k];
  }
  return f;
}
// B (32x16 bf16), element (k,n) at base[k*stride + n]
__device__ inline v16bf load_b_frag(const bf16_t* base, int stride, int lane) {
  const int n = lane & 15;
  const int half = (lane >> 4) & 1;
  v16bf f;
#pragma unroll
  for (int e = 0; e < 16; ++e) {
    const int p = e >> 1, lo = e & 1;
    const int k = half * 16 + 2 * p + lo;
    f[e] = base[k * stride + n];
  }
  return f;
}
// B (32x16 bf16) from row-major [N][K] storage: element (k,n) at base[n*stride + k]
__device__ inline v16bf load_b_frag_kt(const bf16_t* base, int stride, int lane) {
  const int n = lane & 15;
  const int half = (lane >> 4) & 1;
  v16bf f;
#pragma unroll
  for (int e = 0; e < 16; ++e) {
    const int p = e >> 1, lo = e & 1;
    const int k = half * 16 + 2 * p + lo;
    f[e] = base[n * stride + k];
  }
  return f;
}
// f32 A (16x4): lane<16 K={0,1}, lane>=16 K={2,3}
__device__ inline v2f load_a_frag_f32(const float* base, int stride, int lane) {
  const int m = lane & 15;
  const int half = (lane >> 4) & 1;
  v2f f;
  f[0] = base[m * stride + half * 2 + 0];
  f[1] = base[m * stride + half * 2 + 1];
  return f;
}
// f32 B (4x16), element (k,n) at base[k*stride + n]
__device__ inline v2f load_b_frag_f32(const float* base, int stride, int lane) {
  const int n = lane & 15;
  const int half = (lane >> 4) & 1;
  v2f f;
  f[0] = base[(half * 2 + 0) * stride + n];
  f[1] = base[(half * 2 + 1) * stride + n];
  return f;
}

// ---------------------------------------------------------------------------
// bf16 NT GEMM core: C[M,N] = A[M,K] * B[N,K]^T  (f32 accum)
// 256 threads = 8 waves, block tile 128x128, K step 32.
// ---------------------------------------------------------------------------
#define BT 40  // padded LDS stride (bf16 elems); 40*2=80B, 16B aligned

template <class Epi>
__device__ inline void gemm_bf16_core(const bf16_t* __restrict__ A,
                                      const bf16_t* __restrict__ B,
                                      int K, Epi&& epi) {
  __shared__ bf16_t As[128][BT];
  __shared__ bf16_t Bs[128][BT];
  const int tid  = threadIdx.x;
  const int lane = tid & 31;
  const int wv   = tid >> 5;
  const int wm   = wv >> 2;  // 0..1
  const int wn   = wv & 3;   // 0..3
  const int m0   = blockIdx.y * 128;
  const int n0   = blockIdx.x * 128;

  v8f acc[4][2];
#pragma unroll
  for (int i = 0; i < 4; ++i)
#pragma unroll
    for (int j = 0; j < 2; ++j) acc[i][j] = {};

  const int lrow = tid >> 1;
  const int lcol = (tid & 1) * 16;

  for (int k0 = 0; k0 < K; k0 += 32) {
    {
      const uint4* s = reinterpret_cast<const uint4*>(A + (size_t)(m0 + lrow) * K + k0 + lcol);
      uint4* d = reinterpret_cast<uint4*>(&As[lrow][lcol]);
      d[0] = s[0]; d[1] = s[1];
    }
    {
      const uint4* s = reinterpret_cast<const uint4*>(B + (size_t)(n0 + lrow) * K + k0 + lcol);
      uint4* d = reinterpret_cast<uint4*>(&Bs[lrow][lcol]);
      d[0] = s[0]; d[1] = s[1];
    }
    __syncthreads();
    v16bf af[4], bfr[2];
#pragma unroll
    for (int i = 0; i < 4; ++i) af[i] = load_a_frag(&As[wm * 64 + i * 16][0], BT, lane);
#pragma unroll
    for (int j = 0; j < 2; ++j) bfr[j] = load_b_frag_kt(&Bs[wn * 32 + j * 16][0], BT, lane);
#pragma unroll
    for (int i = 0; i < 4; ++i)
#pragma unroll
      for (int j = 0; j < 2; ++j) acc[i][j] = wmma_bf16(af[i], bfr[j], acc[i][j]);
    __syncthreads();
  }

#pragma unroll
  for (int i = 0; i < 4; ++i)
#pragma unroll
    for (int j = 0; j < 2; ++j)
#pragma unroll
      for (int vg = 0; vg < 8; ++vg) {
        int r = m0 + wm * 64 + i * 16 + ((lane < 16) ? vg : vg + 8);
        int c = n0 + wn * 32 + j * 16 + (lane & 15);
        epi(r, c, acc[i][j][vg]);
      }
}

// ---------------------------------------------------------------------------
// f32 NN GEMM core: C[M,N] = A[M,K] * B[K,N]  via v_wmma_f32_16x16x4_f32
// 128 threads = 4 waves, block tile 64x64, K step 16.
// ---------------------------------------------------------------------------
template <class Epi>
__device__ inline void gemm_f32_core(const float* __restrict__ A,
                                     const float* __restrict__ B,
                                     int N, int K, Epi&& epi) {
  __shared__ float As[64][20];
  __shared__ float Bs[16][68];
  const int tid  = threadIdx.x;
  const int lane = tid & 31;
  const int wv   = tid >> 5;
  const int wm   = wv >> 1;
  const int wn   = wv & 1;
  const int m0   = blockIdx.y * 64;
  const int n0   = blockIdx.x * 64;

  v8f acc[2][2];
#pragma unroll
  for (int i = 0; i < 2; ++i)
#pragma unroll
    for (int j = 0; j < 2; ++j) acc[i][j] = {};

  const int arow = tid >> 1, acb = (tid & 1) * 8;
  const int brow = tid >> 3, bcb = (tid & 7) * 8;

  for (int k0 = 0; k0 < K; k0 += 16) {
    {
      const uint4* s = reinterpret_cast<const uint4*>(A + (size_t)(m0 + arow) * K + k0 + acb);
      uint4* d = reinterpret_cast<uint4*>(&As[arow][acb]);
      d[0] = s[0]; d[1] = s[1];
    }
    {
      const uint4* s = reinterpret_cast<const uint4*>(B + (size_t)(k0 + brow) * N + n0 + bcb);
      uint4* d = reinterpret_cast<uint4*>(&Bs[brow][bcb]);
      d[0] = s[0]; d[1] = s[1];
    }
    __syncthreads();
#pragma unroll
    for (int kk = 0; kk < 16; kk += 4) {
      v2f af[2], bfr[2];
#pragma unroll
      for (int i = 0; i < 2; ++i) af[i] = load_a_frag_f32(&As[wm * 32 + i * 16][kk], 20, lane);
#pragma unroll
      for (int j = 0; j < 2; ++j) bfr[j] = load_b_frag_f32(&Bs[kk][wn * 32 + j * 16], 68, lane);
#pragma unroll
      for (int i = 0; i < 2; ++i)
#pragma unroll
        for (int j = 0; j < 2; ++j) acc[i][j] = wmma_f32(af[i], bfr[j], acc[i][j]);
    }
    __syncthreads();
  }

#pragma unroll
  for (int i = 0; i < 2; ++i)
#pragma unroll
    for (int j = 0; j < 2; ++j)
#pragma unroll
      for (int vg = 0; vg < 8; ++vg) {
        int r = m0 + wm * 32 + i * 16 + ((lane < 16) ? vg : vg + 8);
        int c = n0 + wn * 32 + j * 16 + (lane & 15);
        epi(r, c, acc[i][j][vg]);
      }
}

// ---------------------------------------------------------------------------
// Small utility kernels
// ---------------------------------------------------------------------------
__global__ __launch_bounds__(256) void k_cvt_bf16(bf16_t* __restrict__ dst,
                                                  const float* __restrict__ src, int n) {
  int i = blockIdx.x * 256 + threadIdx.x;
  if (i < n) dst[i] = (bf16_t)src[i];
}

__global__ __launch_bounds__(1) void k_zero_u32(unsigned* p) { *p = 0u; }

__global__ __launch_bounds__(128) void k_layernorm(const float* __restrict__ x,
                                                   const float* __restrict__ w,
                                                   const float* __restrict__ b,
                                                   bf16_t* __restrict__ h) {
  const int row = blockIdx.x, t = threadIdx.x;
  __shared__ float red[128];
  const float* xr = x + (size_t)row * 512;
  float vals[4];
  float s = 0.f;
#pragma unroll
  for (int i = 0; i < 4; ++i) { vals[i] = xr[t + i * 128]; s += vals[i]; }
  red[t] = s; __syncthreads();
  for (int off = 64; off > 0; off >>= 1) { if (t < off) red[t] += red[t + off]; __syncthreads(); }
  const float mu = red[0] * (1.f / 512.f);
  __syncthreads();
  float sq = 0.f;
#pragma unroll
  for (int i = 0; i < 4; ++i) { float d = vals[i] - mu; sq += d * d; }
  red[t] = sq; __syncthreads();
  for (int off = 64; off > 0; off >>= 1) { if (t < off) red[t] += red[t + off]; __syncthreads(); }
  const float rstd = rsqrtf(red[0] * (1.f / 512.f) + 1e-5f);
#pragma unroll
  for (int i = 0; i < 4; ++i) {
    int c = t + i * 128;
    h[(size_t)row * 512 + c] = (bf16_t)((vals[i] - mu) * rstd * w[c] + b[c]);
  }
}

// landmark mean pooling: q_l/k_l[h][m][d] = mean over 64 tokens
__global__ __launch_bounds__(64) void k_pool(const bf16_t* __restrict__ q,
                                             const bf16_t* __restrict__ k,
                                             bf16_t* __restrict__ ql,
                                             bf16_t* __restrict__ kl) {
  const int h = blockIdx.y, m = blockIdx.x, d = threadIdx.x;
  size_t base = ((size_t)h * 16384 + (size_t)m * 64) * 64 + d;
  float sq = 0.f, sk = 0.f;
#pragma unroll 8
  for (int i = 0; i < 64; ++i) { sq += (float)q[base + (size_t)i * 64]; sk += (float)k[base + (size_t)i * 64]; }
  ql[((size_t)h * 256 + m) * 64 + d] = (bf16_t)(sq * (1.f / 64.f));
  kl[((size_t)h * 256 + m) * 64 + d] = (bf16_t)(sk * (1.f / 64.f));
}

// row softmax, rows of length 256 (attn2)
__global__ __launch_bounds__(256) void k_softmax256(float* __restrict__ a) {
  const size_t row = blockIdx.x;
  const int t = threadIdx.x;
  __shared__ float red[256];
  float v = a[row * 256 + t];
  red[t] = v; __syncthreads();
  for (int off = 128; off > 0; off >>= 1) { if (t < off) red[t] = fmaxf(red[t], red[t + off]); __syncthreads(); }
  const float mx = red[0]; __syncthreads();
  float e = __expf(v - mx);
  red[t] = e; __syncthreads();
  for (int off = 128; off > 0; off >>= 1) { if (t < off) red[t] += red[t + off]; __syncthreads(); }
  a[row * 256 + t] = e / red[0];
}

// max column-abs-sum over all heads (row sums are exactly 1 post-softmax)
__global__ __launch_bounds__(256) void k_colmax(const float* __restrict__ a, unsigned* mx) {
  const int h = blockIdx.x, j = threadIdx.x;
  __shared__ float red[256];
  float s = 0.f;
  for (int i = 0; i < 256; ++i) s += fabsf(a[((size_t)h * 256 + i) * 256 + j]);
  red[j] = s; __syncthreads();
  for (int off = 128; off > 0; off >>= 1) { if (j < off) red[j] = fmaxf(red[j], red[j + off]); __syncthreads(); }
  if (j == 0) atomicMax(mx, __float_as_uint(red[0]));
}

// z0 = a^T / (1 * colmax)
__global__ __launch_bounds__(256) void k_pinv_init(const float* __restrict__ a,
                                                   float* __restrict__ z,
                                                   const unsigned* mx) {
  size_t idx = (size_t)blockIdx.x * 256 + threadIdx.x;
  int j = (int)(idx & 255), i = (int)((idx >> 8) & 255), h = (int)(idx >> 16);
  float denom = __uint_as_float(*mx);
  z[idx] = a[((size_t)h * 256 + j) * 256 + i] / denom;
}

// u = d*I - src  over [8][256][256]
__global__ __launch_bounds__(256) void k_isub(float* __restrict__ u,
                                              const float* __restrict__ src, float dval) {
  size_t idx = (size_t)blockIdx.x * 256 + threadIdx.x;
  int j = (int)(idx & 255), i = (int)((idx >> 8) & 255);
  u[idx] = ((i == j) ? dval : 0.f) - src[idx];
}

// ---------------------------------------------------------------------------
// GEMM wrapper kernels
// ---------------------------------------------------------------------------
__global__ __launch_bounds__(256) void k_gemm_qkv(const bf16_t* __restrict__ A,
                                                  const bf16_t* __restrict__ B,
                                                  bf16_t* __restrict__ q,
                                                  bf16_t* __restrict__ k,
                                                  bf16_t* __restrict__ v) {
  gemm_bf16_core(A, B, 512, [=](int r, int c, float val) {
    int sec = c >> 9, cc = c & 511, hd = cc >> 6, dd = cc & 63;
    bf16_t* dst = (sec == 0) ? q : (sec == 1) ? k : v;
    float s = (sec == 0) ? 0.125f : 1.0f;  // q *= DHEAD^-0.5
    dst[((size_t)hd * 16384 + r) * 64 + dd] = (bf16_t)(val * s);
  });
}

__global__ __launch_bounds__(256) void k_gemm_attn2(const bf16_t* __restrict__ ql,
                                                    const bf16_t* __restrict__ kl,
                                                    float* __restrict__ c) {
  const bf16_t* A = ql + (size_t)blockIdx.z * 256 * 64;
  const bf16_t* B = kl + (size_t)blockIdx.z * 256 * 64;
  float* C = c + (size_t)blockIdx.z * 65536;
  gemm_bf16_core(A, B, 64, [=](int r, int cc, float v) { C[(size_t)r * 256 + cc] = v; });
}

__global__ __launch_bounds__(256) void k_gemm_out(const bf16_t* __restrict__ A,
                                                  const bf16_t* __restrict__ B,
                                                  const float* __restrict__ bias,
                                                  const float* __restrict__ x,
                                                  float* __restrict__ out) {
  gemm_bf16_core(A, B, 512, [=](int r, int c, float val) {
    size_t idx = (size_t)r * 512 + c;
    out[idx] = val + bias[c] + x[idx];
  });
}

__global__ __launch_bounds__(128) void k_gemm_f32(const float* __restrict__ A,
                                                  const float* __restrict__ B,
                                                  float* __restrict__ C,
                                                  int N, int K, float scale,
                                                  long sA, long sB, long sC) {
  const float* Ab = A + (size_t)blockIdx.z * sA;
  const float* Bb = B + (size_t)blockIdx.z * sB;
  float* Cb = C + (size_t)blockIdx.z * sC;
  gemm_f32_core(Ab, Bb, N, K, [=](int r, int c, float v) { Cb[(size_t)r * N + c] = v * scale; });
}

// ---------------------------------------------------------------------------
// Fused attn3 @ v with online (flash) softmax over N=16384 keys.
// Grid (16, 8), 128 threads (4 waves). Each block: 16 landmark rows of a head.
// ---------------------------------------------------------------------------
__global__ __launch_bounds__(128) void k_attn3v(const bf16_t* __restrict__ ql,
                                                const bf16_t* __restrict__ kk,
                                                const bf16_t* __restrict__ vv,
                                                float* __restrict__ t1) {
  const int h = blockIdx.y;
  const int m0 = blockIdx.x * 16;
  const int tid = threadIdx.x;
  const int lane = tid & 31;
  const int wv = tid >> 5;

  __shared__ bf16_t qs[16][72];
  __shared__ bf16_t ks[64][72];
  __shared__ bf16_t vs[64][72];
  __shared__ float sbuf[16][64];
  __shared__ bf16_t pbuf[16][72];
  __shared__ float red[16][8];
  __shared__ float mrow[16], lrowS[16], arow[16];

  {
    int r = tid >> 3, cb = (tid & 7) * 8;
    *(uint4*)&qs[r][cb] = *(const uint4*)(ql + ((size_t)h * 256 + m0 + r) * 64 + cb);
  }
  if (tid < 16) { mrow[tid] = -1e30f; lrowS[tid] = 0.f; }
  v8f acc = {};
  __syncthreads();

  const int r8 = tid >> 3, l8 = tid & 7;

  for (int kt = 0; kt < 16384; kt += 64) {
    {
      int r = tid >> 1, cb = (tid & 1) * 32;
      const uint4* s = (const uint4*)(kk + ((size_t)h * 16384 + kt + r) * 64 + cb);
      uint4* d = (uint4*)&ks[r][cb];
      d[0] = s[0]; d[1] = s[1]; d[2] = s[2]; d[3] = s[3];
      const uint4* s2 = (const uint4*)(vv + ((size_t)h * 16384 + kt + r) * 64 + cb);
      uint4* d2 = (uint4*)&vs[r][cb];
      d2[0] = s2[0]; d2[1] = s2[1]; d2[2] = s2[2]; d2[3] = s2[3];
    }
    __syncthreads();

    // S[:, wv*16 .. +16) = qs @ ks^T
    v8f s = {};
#pragma unroll
    for (int k2 = 0; k2 < 2; ++k2) {
      v16bf a = load_a_frag(&qs[0][k2 * 32], 72, lane);
      v16bf b = load_b_frag_kt(&ks[wv * 16][k2 * 32], 72, lane);
      s = wmma_bf16(a, b, s);
    }
#pragma unroll
    for (int vg = 0; vg < 8; ++vg) {
      int rr = (lane < 16) ? vg : vg + 8;
      sbuf[rr][wv * 16 + (lane & 15)] = s[vg];
    }
    __syncthreads();

    // online softmax update
    float mx = -1e30f;
#pragma unroll
    for (int j = 0; j < 8; ++j) mx = fmaxf(mx, sbuf[r8][l8 + j * 8]);
    red[r8][l8] = mx;
    __syncthreads();
    if (l8 == 0) {
      float m2 = red[r8][0];
      for (int j = 1; j < 8; ++j) m2 = fmaxf(m2, red[r8][j]);
      float mo = mrow[r8];
      float mn = fmaxf(mo, m2);
      arow[r8] = __expf(mo - mn);
      mrow[r8] = mn;
    }
    __syncthreads();
    const float mn = mrow[r8];
    float ps = 0.f;
#pragma unroll
    for (int j = 0; j < 8; ++j) {
      float p = __expf(sbuf[r8][l8 + j * 8] - mn);
      pbuf[r8][l8 + j * 8] = (bf16_t)p;
      ps += p;
    }
    red[r8][l8] = ps;
    __syncthreads();
    if (l8 == 0) {
      float s2 = 0.f;
      for (int j = 0; j < 8; ++j) s2 += red[r8][j];
      lrowS[r8] = lrowS[r8] * arow[r8] + s2;
    }
    __syncthreads();

    // rescale accumulator, then acc += P @ V
#pragma unroll
    for (int vg = 0; vg < 8; ++vg) {
      int rr = (lane < 16) ? vg : vg + 8;
      acc[vg] *= arow[rr];
    }
#pragma unroll
    for (int k2 = 0; k2 < 2; ++k2) {
      v16bf a = load_a_frag(&pbuf[0][k2 * 32], 72, lane);
      v16bf b = load_b_frag(&vs[k2 * 32][wv * 16], 72, lane);
      acc = wmma_bf16(a, b, acc);
    }
    __syncthreads();
  }
#pragma unroll
  for (int vg = 0; vg < 8; ++vg) {
    int rr = (lane < 16) ? vg : vg + 8;
    t1[((size_t)h * 256 + m0 + rr) * 64 + wv * 16 + (lane & 15)] = acc[vg] / lrowS[rr];
  }
}

// ---------------------------------------------------------------------------
// Fused attn1 = softmax(q @ k_l^T) then @ t2 -> out_heads (f32).
// Grid (1024, 8), 128 threads (4 waves). 16 query rows/block.
// ---------------------------------------------------------------------------
__global__ __launch_bounds__(128) void k_attn1_out(const bf16_t* __restrict__ q,
                                                   const bf16_t* __restrict__ kl,
                                                   const bf16_t* __restrict__ t2,
                                                   float* __restrict__ oh) {
  const int h = blockIdx.y;
  const int n0b = blockIdx.x * 16;
  const int tid = threadIdx.x;
  const int lane = tid & 31;
  const int wv = tid >> 5;

  __shared__ bf16_t qs[16][72];
  __shared__ float sbuf[16][264];
  __shared__ bf16_t pbuf[16][264];
  __shared__ float red[16][8];
  __shared__ float mrow[16], lrowS[16];

  {
    int r = tid >> 3, cb = (tid & 7) * 8;
    *(uint4*)&qs[r][cb] = *(const uint4*)(q + ((size_t)h * 16384 + n0b + r) * 64 + cb);
  }
  __syncthreads();

  const bf16_t* klh = kl + (size_t)h * 256 * 64;
#pragma unroll
  for (int j = 0; j < 4; ++j) {
    v8f s = {};
#pragma unroll
    for (int k2 = 0; k2 < 2; ++k2) {
      v16bf a = load_a_frag(&qs[0][k2 * 32], 72, lane);
      v16bf b = load_b_frag_kt(klh + (size_t)(wv * 64 + j * 16) * 64 + k2 * 32, 64, lane);
      s = wmma_bf16(a, b, s);
    }
#pragma unroll
    for (int vg = 0; vg < 8; ++vg) {
      int rr = (lane < 16) ? vg : vg + 8;
      sbuf[rr][wv * 64 + j * 16 + (lane & 15)] = s[vg];
    }
  }
  __syncthreads();

  const int r8 = tid >> 3, l8 = tid & 7;
  float mx = -1e30f;
#pragma unroll 8
  for (int j = 0; j < 32; ++j) mx = fmaxf(mx, sbuf[r8][l8 + j * 8]);
  red[r8][l8] = mx;
  __syncthreads();
  if (l8 == 0) {
    float m2 = red[r8][0];
    for (int j = 1; j < 8; ++j) m2 = fmaxf(m2, red[r8][j]);
    mrow[r8] = m2;
  }
  __syncthreads();
  const float mn = mrow[r8];
  float ps = 0.f;
#pragma unroll 8
  for (int j = 0; j < 32; ++j) {
    float p = __expf(sbuf[r8][l8 + j * 8] - mn);
    pbuf[r8][l8 + j * 8] = (bf16_t)p;
    ps += p;
  }
  red[r8][l8] = ps;
  __syncthreads();
  if (l8 == 0) {
    float s2 = 0.f;
    for (int j = 0; j < 8; ++j) s2 += red[r8][j];
    lrowS[r8] = s2;
  }
  __syncthreads();

  const bf16_t* t2h = t2 + (size_t)h * 256 * 64;
  v8f acc = {};
#pragma unroll
  for (int k2 = 0; k2 < 8; ++k2) {
    v16bf a = load_a_frag(&pbuf[0][k2 * 32], 264, lane);
    v16bf b = load_b_frag(t2h + (size_t)(k2 * 32) * 64 + wv * 16, 64, lane);
    acc = wmma_bf16(a, b, acc);
  }
#pragma unroll
  for (int vg = 0; vg < 8; ++vg) {
    int rr = (lane < 16) ? vg : vg + 8;
    oh[((size_t)h * 16384 + n0b + rr) * 64 + wv * 16 + (lane & 15)] = acc[vg] / lrowS[rr];
  }
}

// depthwise 33-tap conv along sequence + concat heads -> bf16 [N,512]
__global__ __launch_bounds__(256) void k_conv_concat(const float* __restrict__ oh,
                                                     const bf16_t* __restrict__ v,
                                                     const float* __restrict__ cw,
                                                     bf16_t* __restrict__ cat) {
  const int h = blockIdx.y;
  const size_t flat = (size_t)blockIdx.x * 256 + threadIdx.x;  // over N*64
  const int n = (int)(flat >> 6);
  const int d = (int)(flat & 63);
  float s = oh[((size_t)h * 16384 + n) * 64 + d];
#pragma unroll
  for (int j = 0; j < 33; ++j) {
    int nn = n + j - 16;
    if (nn >= 0 && nn < 16384)
      s += cw[h * 33 + j] * (float)v[((size_t)h * 16384 + nn) * 64 + d];
  }
  cat[(size_t)n * 512 + h * 64 + d] = (bf16_t)s;
}

// ---------------------------------------------------------------------------
// Host orchestration
// ---------------------------------------------------------------------------
extern "C" void kernel_launch(void* const* d_in, const int* in_sizes, int n_in,
                              void* d_out, int out_size, void* d_ws, size_t ws_size,
                              hipStream_t stream) {
  (void)in_sizes; (void)n_in; (void)out_size; (void)ws_size;
  const float* x      = (const float*)d_in[0];
  const float* norm_w = (const float*)d_in[1];
  const float* norm_b = (const float*)d_in[2];
  const float* w_qkv  = (const float*)d_in[3];
  const float* w_out  = (const float*)d_in[4];
  const float* b_out  = (const float*)d_in[5];
  const float* conv_w = (const float*)d_in[6];
  float* out = (float*)d_out;

  const int N = 16384, H = 8, DH = 64, M = 256;
  char* ws = (char*)d_ws;
  size_t off = 0;
  auto alloc = [&](size_t bytes) -> char* {
    char* p = ws + off;
    off = (off + bytes + 255) & ~(size_t)255;
    return p;
  };
  bf16_t* hbf   = (bf16_t*)alloc((size_t)N * 512 * 2);
  bf16_t* wqkvb = (bf16_t*)alloc((size_t)1536 * 512 * 2);
  bf16_t* woutb = (bf16_t*)alloc((size_t)512 * 512 * 2);
  bf16_t* qb    = (bf16_t*)alloc((size_t)H * N * DH * 2);
  bf16_t* kb    = (bf16_t*)alloc((size_t)H * N * DH * 2);
  bf16_t* vb    = (bf16_t*)alloc((size_t)H * N * DH * 2);
  bf16_t* qlb   = (bf16_t*)alloc((size_t)H * M * DH * 2);
  bf16_t* klb   = (bf16_t*)alloc((size_t)H * M * DH * 2);
  float* a2 = (float*)alloc((size_t)H * M * M * 4);
  float* zA = (float*)alloc((size_t)H * M * M * 4);
  float* zB = (float*)alloc((size_t)H * M * M * 4);
  float* az = (float*)alloc((size_t)H * M * M * 4);
  float* tt = (float*)alloc((size_t)H * M * M * 4);
  float* uu = (float*)alloc((size_t)H * M * M * 4);
  unsigned* cmax = (unsigned*)alloc(256);
  float* t1  = (float*)alloc((size_t)H * M * DH * 4);
  float* t2f = (float*)alloc((size_t)H * M * DH * 4);
  bf16_t* t2b = (bf16_t*)alloc((size_t)H * M * DH * 2);
  float* oh = (float*)alloc((size_t)H * N * DH * 4);
  bf16_t* cat = (bf16_t*)alloc((size_t)N * 512 * 2);

  // 1) weights -> bf16, layernorm
  k_cvt_bf16<<<(1536 * 512) / 256, 256, 0, stream>>>(wqkvb, w_qkv, 1536 * 512);
  k_cvt_bf16<<<(512 * 512) / 256, 256, 0, stream>>>(woutb, w_out, 512 * 512);
  k_layernorm<<<N, 128, 0, stream>>>(x, norm_w, norm_b, hbf);

  // 2) qkv projection (WMMA bf16), head-split, q pre-scaled
  k_gemm_qkv<<<dim3(1536 / 128, N / 128), 256, 0, stream>>>(hbf, wqkvb, qb, kb, vb);

  // 3) landmark pooling; attn2 logits + softmax
  k_pool<<<dim3(M, H), 64, 0, stream>>>(qb, kb, qlb, klb);
  k_gemm_attn2<<<dim3(2, 2, H), 256, 0, stream>>>(qlb, klb, a2);
  k_softmax256<<<H * M, 256, 0, stream>>>(a2);

  // 4) Moore-Penrose pinv (f32 WMMA)
  k_zero_u32<<<1, 1, 0, stream>>>(cmax);
  k_colmax<<<H, 256, 0, stream>>>(a2, cmax);
  k_pinv_init<<<(H * M * M) / 256, 256, 0, stream>>>(a2, zA, cmax);
  float* zc = zA; float* zn = zB;
  for (int it = 0; it < 6; ++it) {
    k_gemm_f32<<<dim3(4, 4, H), 128, 0, stream>>>(a2, zc, az, 256, 256, 1.f, 65536, 65536, 65536);
    k_isub<<<(H * M * M) / 256, 256, 0, stream>>>(uu, az, 7.f);
    k_gemm_f32<<<dim3(4, 4, H), 128, 0, stream>>>(az, uu, tt, 256, 256, 1.f, 65536, 65536, 65536);
    k_isub<<<(H * M * M) / 256, 256, 0, stream>>>(uu, tt, 15.f);
    k_gemm_f32<<<dim3(4, 4, H), 128, 0, stream>>>(az, uu, tt, 256, 256, 1.f, 65536, 65536, 65536);
    k_isub<<<(H * M * M) / 256, 256, 0, stream>>>(uu, tt, 13.f);
    k_gemm_f32<<<dim3(4, 4, H), 128, 0, stream>>>(zc, uu, zn, 256, 256, 0.25f, 65536, 65536, 65536);
    float* tmp = zc; zc = zn; zn = tmp;
  }

  // 5) t1 = softmax(attn3) @ v  (flash, online softmax)
  k_attn3v<<<dim3(M / 16, H), 128, 0, stream>>>(qlb, kb, vb, t1);

  // 6) t2 = z @ t1 (f32 WMMA), convert to bf16
  k_gemm_f32<<<dim3(1, 4, H), 128, 0, stream>>>(zc, t1, t2f, 64, 256, 1.f, 65536, 16384, 16384);
  k_cvt_bf16<<<(H * M * DH) / 256, 256, 0, stream>>>(t2b, t2f, H * M * DH);

  // 7) out_heads = softmax(attn1) @ t2 (fused)
  k_attn1_out<<<dim3(N / 16, H), 128, 0, stream>>>(qb, klb, t2b, oh);

  // 8) depthwise conv residual + concat heads
  k_conv_concat<<<dim3((N * DH) / 256, H), 256, 0, stream>>>(oh, vb, conv_w, cat);

  // 9) final projection + bias + residual (WMMA bf16)
  k_gemm_out<<<dim3(512 / 128, N / 128), 256, 0, stream>>>(cat, woutb, b_out, x, out);
}
